// BlockMoE_16999480557829
// MI455X (gfx1250) — compile-verified, matrix operand order checked by reading, not run on hardware
//
#include <hip/hip_runtime.h>
#include <hip/hip_bf16.h>
#include <math.h>

#define Tn  4096
#define Dd  1024
#define Hh  16
#define DKk 64
#define Ei  8
#define HIDn 4096

typedef _Float16 v8h  __attribute__((ext_vector_type(8)));
typedef _Float16 v16h __attribute__((ext_vector_type(16)));
typedef float    v8f  __attribute__((ext_vector_type(8)));
typedef int      v4i  __attribute__((ext_vector_type(4)));

#if __has_builtin(__builtin_amdgcn_global_load_async_to_lds_b128) && \
    __has_builtin(__builtin_amdgcn_s_wait_asynccnt)
#define HAVE_ASYNC 1
#else
#define HAVE_ASYNC 0
#endif

#if HAVE_ASYNC
typedef __attribute__((address_space(1))) v4i gv4i;  // global (AS1) int4
typedef __attribute__((address_space(3))) v4i lv4i;  // LDS (AS3) int4
// 16B per lane, global -> LDS, no VGPR round trip; tracked by ASYNCcnt
__device__ __forceinline__ void async_cp16(const _Float16* g, _Float16* l) {
  __builtin_amdgcn_global_load_async_to_lds_b128((gv4i*)g, (lv4i*)l, 0, 0);
}
#define ASYNC_WAIT(n) __builtin_amdgcn_s_wait_asynccnt(n)
#else
__device__ __forceinline__ void async_cp16(const _Float16* g, _Float16* l) {
  *(v8h*)l = *(const v8h*)g;
}
#define ASYNC_WAIT(n) do {} while (0)
#endif

__device__ __forceinline__ v8f zf8() {
  v8f z;
#pragma unroll
  for (int i = 0; i < 8; ++i) z[i] = 0.0f;
  return z;
}

__device__ __forceinline__ v8f wmma_f16(v16h a, v16h b, v8f c) {
  // D = A(16x32 f16) * B(32x16 f16) + C(f32)
  return __builtin_amdgcn_wmma_f32_16x16x32_f16(false, a, false, b, (short)0, c,
                                                false, false);
}

// Load a 16-bit operand fragment from LDS laid out [row][K] with `stride`
// elements per row. CDNA5 16-bit A/B layout: lane (l&15) selects row,
// (l>>4) selects K-halves {0-7,16-23} vs {8-15,24-31} within the 32-wide K.
__device__ __forceinline__ v16h ld_frag(const _Float16* base, int row,
                                        int stride, int k0, int lane) {
  const int hi = lane >> 4;
  const _Float16* p = base + row * stride + k0 + hi * 8;
  v8h lo = *(const v8h*)(p);
  v8h hv = *(const v8h*)(p + 16);
  v16h f;
#pragma unroll
  for (int j = 0; j < 8; ++j) { f[j] = lo[j]; f[8 + j] = hv[j]; }
  return f;
}

#define LDS_STR 72  // 64 K-elems + 8 pad -> 144B row stride (16B aligned)

// Issue one 128x64 A-tile + 128x64 B-tile stage: 8 async b128 ops per wave.
template <typename RowMap>
__device__ __forceinline__ void issue_stage(const _Float16* __restrict__ A,
                                            long ldA,
                                            const _Float16* __restrict__ Bt,
                                            long ldB, int k0, int colBase,
                                            RowMap rowmap, _Float16* As,
                                            _Float16* Bs, int tid) {
#pragma unroll
  for (int v = 0; v < 4; ++v) {
    int idx = tid + v * 256;
    int r = idx >> 3, c = (idx & 7) * 8;
    long gr = rowmap(r);  // always a valid row (invalid rows masked at store)
    async_cp16(A + gr * ldA + k0 + c, As + r * LDS_STR + c);
    async_cp16(Bt + ((long)colBase + r) * ldB + k0 + c, Bs + r * LDS_STR + c);
  }
}

// One 128x128x64 tile of WMMAs from statically-named LDS buffers (-> ds_load).
__device__ __forceinline__ void gemm_compute(const _Float16* Asl,
                                             const _Float16* Bsl,
                                             v8f acc[4][2], int lane, int wm,
                                             int wn) {
#pragma unroll
  for (int ks = 0; ks < 2; ++ks) {
    v16h af[4], bf[2];
#pragma unroll
    for (int i = 0; i < 4; ++i)
      af[i] = ld_frag(Asl, wm * 64 + i * 16 + (lane & 15), LDS_STR, ks * 32, lane);
#pragma unroll
    for (int j = 0; j < 2; ++j)
      bf[j] = ld_frag(Bsl, wn * 32 + j * 16 + (lane & 15), LDS_STR, ks * 32, lane);
#pragma unroll
    for (int i = 0; i < 4; ++i)
#pragma unroll
      for (int j = 0; j < 2; ++j) acc[i][j] = wmma_f16(af[i], bf[j], acc[i][j]);
  }
}

// Double-buffered WMMA mainloop, pipeline unrolled by 2 so each compute stage
// references a compile-time-known LDS buffer (keeps ds_load, avoids flat).
// Async in-order completion: after issuing the next stage's 8 ops,
// s_wait_asynccnt 8 guarantees the current stage's 8 ops are resident.
// S = K/64 is even for every GEMM in this pipeline.
template <typename RowMap>
__device__ __forceinline__ void gemm_mainloop(
    const _Float16* __restrict__ A, long ldA, const _Float16* __restrict__ Bt,
    long ldB, int K, int colBase, RowMap rowmap, _Float16* As0, _Float16* Bs0,
    _Float16* As1, _Float16* Bs1, v8f acc[4][2], int tid) {
  const int lane = tid & 31, wave = tid >> 5;
  const int wm = wave >> 2, wn = wave & 3;
  const int S = K / 64;
  issue_stage(A, ldA, Bt, ldB, 0, colBase, rowmap, As0, Bs0, tid);
  for (int s = 0; s < S; s += 2) {
    issue_stage(A, ldA, Bt, ldB, (s + 1) * 64, colBase, rowmap, As1, Bs1, tid);
    ASYNC_WAIT(8);
    __syncthreads();
    gemm_compute(As0, Bs0, acc, lane, wm, wn);
    __syncthreads();
    if (s + 2 < S) {
      issue_stage(A, ldA, Bt, ldB, (s + 2) * 64, colBase, rowmap, As0, Bs0, tid);
      ASYNC_WAIT(8);
    } else {
      ASYNC_WAIT(0);
    }
    __syncthreads();
    gemm_compute(As1, Bs1, acc, lane, wm, wn);
    __syncthreads();
  }
}

// ---------------- utility kernels ----------------

__global__ __launch_bounds__(32) void k_zero_counts(int* counts) {
  if (threadIdx.x < Ei) counts[threadIdx.x] = 0;
}

// out[C][R] (f16) = transpose(in[R][C] (f32)); batched over blockIdx.z
__global__ __launch_bounds__(256) void k_transpose_cast(
    const float* __restrict__ in, _Float16* __restrict__ out, int R, int C) {
  __shared__ float tile[32][33];
  const size_t bo = (size_t)blockIdx.z * R * C;
  const int c0 = blockIdx.x * 32, r0 = blockIdx.y * 32;
  for (int i = threadIdx.y; i < 32; i += 8)
    tile[i][threadIdx.x] = in[bo + (size_t)(r0 + i) * C + c0 + threadIdx.x];
  __syncthreads();
  for (int i = threadIdx.y; i < 32; i += 8)
    out[bo + (size_t)(c0 + i) * R + r0 + threadIdx.x] =
        (_Float16)tile[threadIdx.x][i];
}

__global__ __launch_bounds__(256) void k_layernorm(
    const float* __restrict__ X, const float* __restrict__ w,
    const float* __restrict__ b, _Float16* __restrict__ out) {
  __shared__ float r1[256], r2[256];
  const int t = blockIdx.x, tid = threadIdx.x;
  const float* row = X + (size_t)t * Dd;
  float v[4], s = 0.f, s2 = 0.f;
#pragma unroll
  for (int i = 0; i < 4; ++i) {
    v[i] = row[tid * 4 + i];
    s += v[i];
    s2 += v[i] * v[i];
  }
  r1[tid] = s; r2[tid] = s2;
  __syncthreads();
  for (int st = 128; st > 0; st >>= 1) {
    if (tid < st) { r1[tid] += r1[tid + st]; r2[tid] += r2[tid + st]; }
    __syncthreads();
  }
  const float mean = r1[0] * (1.0f / Dd);
  const float var = r2[0] * (1.0f / Dd) - mean * mean;
  const float rstd = rsqrtf(var + 1e-5f);
#pragma unroll
  for (int i = 0; i < 4; ++i) {
    int d = tid * 4 + i;
    out[(size_t)t * Dd + d] = (_Float16)((v[i] - mean) * rstd * w[d] + b[d]);
  }
}

// full-f32 gate: recompute LN2 + logits + first-max argmax (router must be exact)
__global__ __launch_bounds__(128) void k_gate(
    const float* __restrict__ X1, const float* __restrict__ ln2w,
    const float* __restrict__ ln2b, const float* __restrict__ gw,
    const float* __restrict__ gb, int* __restrict__ eid) {
  __shared__ float r1[128], r2[128];
  __shared__ float accs[128][Ei];
  const int t = blockIdx.x, tid = threadIdx.x;
  const float* row = X1 + (size_t)t * Dd;
  float vals[8], s = 0.f, s2 = 0.f;
#pragma unroll
  for (int i = 0; i < 8; ++i) {
    vals[i] = row[tid + i * 128];
    s += vals[i]; s2 += vals[i] * vals[i];
  }
  r1[tid] = s; r2[tid] = s2;
  __syncthreads();
  for (int st = 64; st > 0; st >>= 1) {
    if (tid < st) { r1[tid] += r1[tid + st]; r2[tid] += r2[tid + st]; }
    __syncthreads();
  }
  const float mean = r1[0] * (1.0f / Dd);
  const float var = r2[0] * (1.0f / Dd) - mean * mean;
  const float rstd = rsqrtf(var + 1e-5f);
  float a[Ei];
#pragma unroll
  for (int e = 0; e < Ei; ++e) a[e] = 0.f;
#pragma unroll
  for (int i = 0; i < 8; ++i) {
    int d = tid + i * 128;
    float xn = (vals[i] - mean) * rstd * ln2w[d] + ln2b[d];
#pragma unroll
    for (int e = 0; e < Ei; ++e) a[e] += xn * gw[d * Ei + e];
  }
#pragma unroll
  for (int e = 0; e < Ei; ++e) accs[tid][e] = a[e];
  __syncthreads();
  for (int st = 64; st > 0; st >>= 1) {
    if (tid < st)
#pragma unroll
      for (int e = 0; e < Ei; ++e) accs[tid][e] += accs[tid + st][e];
    __syncthreads();
  }
  if (tid == 0) {
    float best = -3.0e38f; int bi = 0;
#pragma unroll
    for (int e = 0; e < Ei; ++e) {
      float L = accs[0][e] + gb[e];
      if (L > best) { best = L; bi = e; }
    }
    eid[t] = bi;
  }
}

__global__ __launch_bounds__(256) void k_scatter(const int* __restrict__ eid,
                                                 int* counts, int* lists) {
  int t = blockIdx.x * 256 + threadIdx.x;
  if (t < Tn) {
    int e = eid[t];
    int p = atomicAdd(&counts[e], 1);
    lists[e * Tn + p] = t;
  }
}

// ---------------- GEMM kernels ----------------

// qkv = xn1[T,D] @ wqkv  (Bt = wqkv^T [3D][D]); scatter to Q/K/Vt f16
__global__ __launch_bounds__(256) void k_gemm_qkv(
    const _Float16* __restrict__ xn, const _Float16* __restrict__ wT,
    _Float16* __restrict__ Q, _Float16* __restrict__ Kb,
    _Float16* __restrict__ Vt) {
  __shared__ __align__(16) _Float16 As0[128 * LDS_STR], As1[128 * LDS_STR];
  __shared__ __align__(16) _Float16 Bs0[128 * LDS_STR], Bs1[128 * LDS_STR];
  const int tid = threadIdx.x;
  const int rowBase = blockIdx.y * 128, colBase = blockIdx.x * 128;
  v8f acc[4][2];
#pragma unroll
  for (int i = 0; i < 4; ++i)
#pragma unroll
    for (int j = 0; j < 2; ++j) acc[i][j] = zf8();
  auto rm = [&](int r) -> long { return rowBase + r; };
  gemm_mainloop(xn, Dd, wT, Dd, Dd, colBase, rm, As0, Bs0, As1, Bs1, acc, tid);
  const int lane = tid & 31, wave = tid >> 5;
  const int wm = wave >> 2, wn = wave & 3, hi = lane >> 4, ln = lane & 15;
#pragma unroll
  for (int i = 0; i < 4; ++i)
#pragma unroll
    for (int j = 0; j < 2; ++j)
#pragma unroll
      for (int r = 0; r < 8; ++r) {
        int t = rowBase + wm * 64 + i * 16 + r + 8 * hi;
        int col = colBase + wn * 32 + j * 16 + ln;
        int s = col >> 10, rem = col & 1023, h = rem >> 6, dk = rem & 63;
        int bh = (t >> 10) * Hh + h, n = t & 1023;
        _Float16 hv = (_Float16)acc[i][j][r];
        if (s == 0)      Q[((size_t)bh << 16) + (size_t)n * DKk + dk] = hv;
        else if (s == 1) Kb[((size_t)bh << 16) + (size_t)n * DKk + dk] = hv;
        else             Vt[((size_t)bh << 16) + (size_t)dk * 1024 + n] = hv;
      }
}

// x1 = x + Y @ w_proj
__global__ __launch_bounds__(256) void k_gemm_proj(
    const _Float16* __restrict__ Y, const _Float16* __restrict__ wT,
    const float* __restrict__ X, float* __restrict__ X1) {
  __shared__ __align__(16) _Float16 As0[128 * LDS_STR], As1[128 * LDS_STR];
  __shared__ __align__(16) _Float16 Bs0[128 * LDS_STR], Bs1[128 * LDS_STR];
  const int tid = threadIdx.x;
  const int rowBase = blockIdx.y * 128, colBase = blockIdx.x * 128;
  v8f acc[4][2];
#pragma unroll
  for (int i = 0; i < 4; ++i)
#pragma unroll
    for (int j = 0; j < 2; ++j) acc[i][j] = zf8();
  auto rm = [&](int r) -> long { return rowBase + r; };
  gemm_mainloop(Y, Dd, wT, Dd, Dd, colBase, rm, As0, Bs0, As1, Bs1, acc, tid);
  const int lane = tid & 31, wave = tid >> 5;
  const int wm = wave >> 2, wn = wave & 3, hi = lane >> 4, ln = lane & 15;
#pragma unroll
  for (int i = 0; i < 4; ++i)
#pragma unroll
    for (int j = 0; j < 2; ++j)
#pragma unroll
      for (int r = 0; r < 8; ++r) {
        int t = rowBase + wm * 64 + i * 16 + r + 8 * hi;
        int col = colBase + wn * 32 + j * 16 + ln;
        size_t o = (size_t)t * Dd + col;
        X1[o] = X[o] + acc[i][j][r];
      }
}

// MoE layer 1: h = gelu_tanh(gather(xn2) @ w1[e]); grouped by expert lists
__global__ __launch_bounds__(256) void k_moe1(
    const _Float16* __restrict__ xn2, const _Float16* __restrict__ w1T,
    const int* __restrict__ counts, const int* __restrict__ lists,
    _Float16* __restrict__ Hbuf) {
  __shared__ __align__(16) _Float16 As0[128 * LDS_STR], As1[128 * LDS_STR];
  __shared__ __align__(16) _Float16 Bs0[128 * LDS_STR], Bs1[128 * LDS_STR];
  __shared__ int toks[128];
  const int e = blockIdx.y >> 5, chunk = blockIdx.y & 31;
  const int cnt = counts[e], base = chunk * 128;
  if (base >= cnt) return;
  int rem = cnt - base; if (rem > 128) rem = 128;
  const int tid = threadIdx.x;
  if (tid < 128) {
    int idx = (tid < rem) ? tid : (rem - 1);  // clamp: rows >= rem masked at store
    toks[tid] = lists[e * Tn + base + idx];
  }
  __syncthreads();
  const int colBase = blockIdx.x * 128;
  v8f acc[4][2];
#pragma unroll
  for (int i = 0; i < 4; ++i)
#pragma unroll
    for (int j = 0; j < 2; ++j) acc[i][j] = zf8();
  auto rm = [&](int r) -> long { return (long)toks[r]; };
  gemm_mainloop(xn2, Dd, w1T + (size_t)e * HIDn * Dd, Dd, Dd, colBase, rm,
                As0, Bs0, As1, Bs1, acc, tid);
  const int lane = tid & 31, wave = tid >> 5;
  const int wm = wave >> 2, wn = wave & 3, hi = lane >> 4, ln = lane & 15;
#pragma unroll
  for (int i = 0; i < 4; ++i)
#pragma unroll
    for (int j = 0; j < 2; ++j)
#pragma unroll
      for (int r = 0; r < 8; ++r) {
        int rl = wm * 64 + i * 16 + r + 8 * hi;
        if (rl < rem) {
          int tok = toks[rl];
          int col = colBase + wn * 32 + j * 16 + ln;
          float xv = acc[i][j][r];
          float g = 0.5f * xv *
                    (1.0f + tanhf(0.7978845608f * (xv + 0.044715f * xv * xv * xv)));
          Hbuf[(size_t)tok * HIDn + col] = (_Float16)g;
        }
      }
}

// MoE layer 2 + residual: out = x1 + gather(h) @ w2[e]
__global__ __launch_bounds__(256) void k_moe2(
    const _Float16* __restrict__ Hbuf, const _Float16* __restrict__ w2T,
    const int* __restrict__ counts, const int* __restrict__ lists,
    const float* __restrict__ X1, float* __restrict__ Out) {
  __shared__ __align__(16) _Float16 As0[128 * LDS_STR], As1[128 * LDS_STR];
  __shared__ __align__(16) _Float16 Bs0[128 * LDS_STR], Bs1[128 * LDS_STR];
  __shared__ int toks[128];
  const int e = blockIdx.y >> 5, chunk = blockIdx.y & 31;
  const int cnt = counts[e], base = chunk * 128;
  if (base >= cnt) return;
  int rem = cnt - base; if (rem > 128) rem = 128;
  const int tid = threadIdx.x;
  if (tid < 128) {
    int idx = (tid < rem) ? tid : (rem - 1);
    toks[tid] = lists[e * Tn + base + idx];
  }
  __syncthreads();
  const int colBase = blockIdx.x * 128;
  v8f acc[4][2];
#pragma unroll
  for (int i = 0; i < 4; ++i)
#pragma unroll
    for (int j = 0; j < 2; ++j) acc[i][j] = zf8();
  auto rm = [&](int r) -> long { return (long)toks[r]; };
  gemm_mainloop(Hbuf, HIDn, w2T + (size_t)e * Dd * HIDn, HIDn, HIDn, colBase,
                rm, As0, Bs0, As1, Bs1, acc, tid);
  const int lane = tid & 31, wave = tid >> 5;
  const int wm = wave >> 2, wn = wave & 3, hi = lane >> 4, ln = lane & 15;
#pragma unroll
  for (int i = 0; i < 4; ++i)
#pragma unroll
    for (int j = 0; j < 2; ++j)
#pragma unroll
      for (int r = 0; r < 8; ++r) {
        int rl = wm * 64 + i * 16 + r + 8 * hi;
        if (rl < rem) {
          int tok = toks[rl];
          int col = colBase + wn * 32 + j * 16 + ln;
          size_t o = (size_t)tok * Dd + col;
          Out[o] = X1[o] + acc[i][j][r];
        }
      }
}

// ---------------- flash attention ----------------
// 4 waves, 64 query rows per block; each wave owns a 16-row strip.
// K/V tiles double-buffered via async LDS copies (16 async ops/wave/stage),
// pipeline unrolled by 2 so LDS buffers are compile-time known.
#define VSTR 136

__device__ __forceinline__ void attn_issue(const _Float16* Kp,
                                           const _Float16* Vp, int kb,
                                           _Float16* Ks, _Float16* Vs,
                                           int tid) {
  for (int v = tid; v < 1024; v += 128) {
    int r = v >> 3, c = (v & 7) * 8;
    async_cp16(Kp + (size_t)(kb * 128 + r) * DKk + c, Ks + r * LDS_STR + c);
  }
  for (int v = tid; v < 1024; v += 128) {
    int r = v >> 4, c = (v & 15) * 8;
    async_cp16(Vp + (size_t)r * 1024 + kb * 128 + c, Vs + r * VSTR + c);
  }
}

// One key-block: S = Q*K^T, online softmax update, O += P*V.
// Enters after a barrier with Ksb/Vsb resident; exits after a barrier.
__device__ __forceinline__ void attn_stage(const _Float16* Ksb,
                                           const _Float16* Vsb, _Float16* myP,
                                           v16h (&aQ)[2], float (&mrow)[8],
                                           float (&lrow)[8], v8f (&o)[4],
                                           int lane) {
  const int ln = lane & 15, hi = lane >> 4;
  const float scale = 0.125f;  // 1/sqrt(64)
  v8f s[8];
#pragma unroll
  for (int j = 0; j < 8; ++j) s[j] = zf8();
#pragma unroll
  for (int ks = 0; ks < 2; ++ks) {
#pragma unroll
    for (int j = 0; j < 8; ++j) {
      v16h bf = ld_frag(Ksb, j * 16 + ln, LDS_STR, ks * 32, lane);
      s[j] = wmma_f16(aQ[ks], bf, s[j]);
    }
  }
#pragma unroll
  for (int r = 0; r < 8; ++r) {
    float mx = -3.0e38f;
#pragma unroll
    for (int j = 0; j < 8; ++j) mx = fmaxf(mx, s[j][r]);
    mx *= scale;
#pragma unroll
    for (int off = 1; off < 16; off <<= 1)
      mx = fmaxf(mx, __shfl_xor(mx, off, 32));
    float mn = fmaxf(mrow[r], mx);
    float al = __expf(mrow[r] - mn);
    mrow[r] = mn;
    float rs = 0.f;
#pragma unroll
    for (int j = 0; j < 8; ++j) {
      float p = __expf(s[j][r] * scale - mn);
      rs += p;
      myP[(r + 8 * hi) * VSTR + j * 16 + ln] = (_Float16)p;
    }
#pragma unroll
    for (int off = 1; off < 16; off <<= 1) rs += __shfl_xor(rs, off, 32);
    lrow[r] = lrow[r] * al + rs;
#pragma unroll
    for (int d = 0; d < 4; ++d) o[d][r] *= al;
  }
  __syncthreads();
#pragma unroll
  for (int ks = 0; ks < 4; ++ks) {
    v16h ap = ld_frag(myP, ln, VSTR, ks * 32, lane);
#pragma unroll
    for (int d = 0; d < 4; ++d) {
      v16h bf = ld_frag(Vsb, d * 16 + ln, VSTR, ks * 32, lane);
      o[d] = wmma_f16(ap, bf, o[d]);
    }
  }
  __syncthreads();
}

__global__ __launch_bounds__(128) void k_attn(const _Float16* __restrict__ Q,
                                              const _Float16* __restrict__ K,
                                              const _Float16* __restrict__ Vt,
                                              _Float16* __restrict__ Y) {
  __shared__ __align__(16) _Float16 Qs[64 * LDS_STR];
  __shared__ __align__(16) _Float16 Ks0[128 * LDS_STR], Ks1[128 * LDS_STR];
  __shared__ __align__(16) _Float16 Vs0[64 * VSTR], Vs1[64 * VSTR];
  __shared__ __align__(16) _Float16 Ps[4 * 16 * VSTR];
  const int tid = threadIdx.x, lane = tid & 31, wave = tid >> 5;
  const int ln = lane & 15, hi = lane >> 4;
  const int bh = blockIdx.y, qBase = blockIdx.x * 64;
  const _Float16* Qp = Q + ((size_t)bh << 16);
  const _Float16* Kp = K + ((size_t)bh << 16);
  const _Float16* Vp = Vt + ((size_t)bh << 16);
  for (int v = tid; v < 512; v += 128) {
    int r = v >> 3, c = (v & 7) * 8;
    async_cp16(Qp + (size_t)(qBase + r) * DKk + c, Qs + r * LDS_STR + c);
  }
  attn_issue(Kp, Vp, 0, Ks0, Vs0, tid);
  ASYNC_WAIT(16);  // 4 Q ops + 16 stage-0 ops outstanding -> Q resident
  __syncthreads();
  v16h aQ[2];
#pragma unroll
  for (int ks = 0; ks < 2; ++ks)
    aQ[ks] = ld_frag(Qs, wave * 16 + ln, LDS_STR, ks * 32, lane);

  float mrow[8], lrow[8];
  v8f o[4];
#pragma unroll
  for (int r = 0; r < 8; ++r) { mrow[r] = -3.0e38f; lrow[r] = 0.f; }
#pragma unroll
  for (int d = 0; d < 4; ++d) o[d] = zf8();
  _Float16* myP = Ps + wave * 16 * VSTR;

  for (int kb = 0; kb < 8; kb += 2) {
    attn_issue(Kp, Vp, kb + 1, Ks1, Vs1, tid);
    ASYNC_WAIT(16);  // in-order: even stage's 16 ops now resident
    __syncthreads();
    attn_stage(Ks0, Vs0, myP, aQ, mrow, lrow, o, lane);
    if (kb + 2 < 8) {
      attn_issue(Kp, Vp, kb + 2, Ks0, Vs0, tid);
      ASYNC_WAIT(16);
    } else {
      ASYNC_WAIT(0);
    }
    __syncthreads();
    attn_stage(Ks1, Vs1, myP, aQ, mrow, lrow, o, lane);
  }
  const int b = bh >> 4, h = bh & 15;
#pragma unroll
  for (int d = 0; d < 4; ++d)
#pragma unroll
    for (int r = 0; r < 8; ++r) {
      float val = o[d][r] / lrow[r];
      int t = b * 1024 + qBase + wave * 16 + r + 8 * hi;
      int col = h * DKk + d * 16 + ln;
      Y[(size_t)t * Dd + col] = (_Float16)val;
    }
}

// ---------------- host launcher ----------------
extern "C" void kernel_launch(void* const* d_in, const int* in_sizes, int n_in,
                              void* d_out, int out_size, void* d_ws,
                              size_t ws_size, hipStream_t stream) {
  (void)in_sizes; (void)n_in; (void)out_size; (void)ws_size;
  const float* x     = (const float*)d_in[0];
  const float* ln1w  = (const float*)d_in[1];
  const float* ln1b  = (const float*)d_in[2];
  const float* wqkv  = (const float*)d_in[3];
  const float* wproj = (const float*)d_in[4];
  const float* ln2w  = (const float*)d_in[5];
  const float* ln2b  = (const float*)d_in[6];
  const float* gw    = (const float*)d_in[7];
  const float* gb    = (const float*)d_in[8];
  const float* w1    = (const float*)d_in[9];
  const float* w2    = (const float*)d_in[10];
  float* out = (float*)d_out;

  char* wp = (char*)d_ws;
  auto alloc = [&](size_t b) -> char* {
    char* p = wp;
    wp += (b + 255) & ~(size_t)255;
    return p;
  };
  _Float16* xn1    = (_Float16*)alloc((size_t)Tn * Dd * 2);
  _Float16* wqkvT  = (_Float16*)alloc((size_t)3 * Dd * Dd * 2);
  _Float16* wprojT = (_Float16*)alloc((size_t)Dd * Dd * 2);
  _Float16* w1T    = (_Float16*)alloc((size_t)Ei * Dd * HIDn * 2);
  _Float16* w2T    = (_Float16*)alloc((size_t)Ei * Dd * HIDn * 2);
  _Float16* Qb     = (_Float16*)alloc((size_t)Tn * Dd * 2);
  _Float16* Kb     = (_Float16*)alloc((size_t)Tn * Dd * 2);
  _Float16* Vtb    = (_Float16*)alloc((size_t)Tn * Dd * 2);
  _Float16* Yh     = (_Float16*)alloc((size_t)Tn * Dd * 2);
  float*    x1     = (float*)   alloc((size_t)Tn * Dd * 4);
  _Float16* xn2    = (_Float16*)alloc((size_t)Tn * Dd * 2);
  _Float16* Hb     = (_Float16*)alloc((size_t)Tn * HIDn * 2);
  int* eid    = (int*)alloc((size_t)Tn * 4);
  int* counts = (int*)alloc((size_t)Ei * 4);
  int* lists  = (int*)alloc((size_t)Ei * Tn * 4);

  k_zero_counts<<<1, 32, 0, stream>>>(counts);
  k_transpose_cast<<<dim3(3072 / 32, 1024 / 32, 1), dim3(32, 8), 0, stream>>>(
      wqkv, wqkvT, 1024, 3072);
  k_transpose_cast<<<dim3(1024 / 32, 1024 / 32, 1), dim3(32, 8), 0, stream>>>(
      wproj, wprojT, 1024, 1024);
  k_transpose_cast<<<dim3(4096 / 32, 1024 / 32, 8), dim3(32, 8), 0, stream>>>(
      w1, w1T, 1024, 4096);
  k_transpose_cast<<<dim3(1024 / 32, 4096 / 32, 8), dim3(32, 8), 0, stream>>>(
      w2, w2T, 4096, 1024);
  k_layernorm<<<Tn, 256, 0, stream>>>(x, ln1w, ln1b, xn1);
  k_gemm_qkv<<<dim3(24, 32), 256, 0, stream>>>(xn1, wqkvT, Qb, Kb, Vtb);
  k_attn<<<dim3(16, 64), 128, 0, stream>>>(Qb, Kb, Vtb, Yh);
  k_gemm_proj<<<dim3(8, 32), 256, 0, stream>>>(Yh, wprojT, x, x1);
  k_layernorm<<<Tn, 256, 0, stream>>>(x1, ln2w, ln2b, xn2);
  k_gate<<<Tn, 128, 0, stream>>>(x1, ln2w, ln2b, gw, gb, eid);
  k_scatter<<<16, 256, 0, stream>>>(eid, counts, lists);
  k_moe1<<<dim3(32, 256), 256, 0, stream>>>(xn2, w1T, counts, lists, Hb);
  k_moe2<<<dim3(8, 256), 256, 0, stream>>>(Hb, w2T, counts, lists, x1, out);
}
